// ISTFTHead_61083024884253
// MI455X (gfx1250) — matrix operand order repaired and checked by reading, not compiled, not run on hardware
//
#include <hip/hip_runtime.h>
#include <hip/hip_bf16.h>
#include <math.h>

// ---------------------------------------------------------------------------
// ISTFT head for MI455X (gfx1250, wave32, WMMA + TDM).
//   h = x @ W + b                         (GEMM1, bf16x3 WMMA, B via TDM->LDS)
//   a2 = [mag*cos(p) | mag*sin(p)]        (activation pack)
//   frames = a2 @ [Cw;Dw]  (windowed irfft as GEMM2, bf16x3 WMMA)
//   y = gather-overlap-add(frames) / env  (analytic env, no atomics)
// ---------------------------------------------------------------------------

typedef __attribute__((ext_vector_type(16))) __bf16    v16bf;
typedef __attribute__((ext_vector_type(8)))  float     v8f;
typedef __attribute__((ext_vector_type(4)))  unsigned  v4u;
typedef __attribute__((ext_vector_type(8)))  int       v8i;
typedef __attribute__((ext_vector_type(4)))  int       v4i;

#if defined(__has_builtin)
#if __has_builtin(__builtin_amdgcn_tensor_load_to_lds) && \
    __has_builtin(__builtin_amdgcn_s_wait_tensorcnt)
#define USE_TDM 1
#endif
#endif
#ifndef USE_TDM
#define USE_TDM 0
#endif

// Problem constants
constexpr int BATCH   = 8;
constexpr int LFRAMES = 2048;
constexpr int DFEAT   = 512;
constexpr int NFFT    = 1024;
constexpr int NBINS   = 513;          // NFFT/2 + 1
constexpr int OCOLS   = NFFT + 2;     // 1026
constexpr int HOP     = 256;
constexpr int PADC    = 384;          // (1024-256)/2
constexpr int OUTLEN  = 524288;       // per-batch output length (2^19)

constexpr int M       = BATCH * LFRAMES;   // 16384
constexpr int MT      = M / 16;            // 1024 M tiles
constexpr int KT1     = DFEAT / 32;        // 16 K tiles (GEMM1)
constexpr int NT1     = 65;                // ceil(1026/16) N tiles (GEMM1)
constexpr int NP1     = NT1 * 16;          // 1040 padded h stride
constexpr int K2      = 1056;              // pad 1026 -> 33*32
constexpr int KT2     = K2 / 32;           // 33 K tiles (GEMM2)
constexpr int NT2     = NFFT / 16;         // 64 N tiles (GEMM2)

constexpr float TWO_PI_OVER_N = 6.28318530717958647692f / 1024.0f;

// Workspace layout (bytes)
constexpr size_t OFF_XHI = 0;
constexpr size_t SZ_XPK  = (size_t)M * DFEAT * 2;              // 16.78 MB
constexpr size_t OFF_XLO = OFF_XHI + SZ_XPK;
constexpr size_t OFF_WHI = OFF_XLO + SZ_XPK;
constexpr size_t SZ_WPK  = (size_t)DFEAT * NP1 * 2;            // 1.06 MB
constexpr size_t OFF_WLO = OFF_WHI + SZ_WPK;
constexpr size_t OFF_H   = OFF_WLO + SZ_WPK;
constexpr size_t SZ_H    = (size_t)M * NP1 * 4;                // 68.2 MB
constexpr size_t OFF_AHI = OFF_H + SZ_H;
constexpr size_t SZ_APK  = (size_t)M * K2 * 2;                 // 34.6 MB
constexpr size_t OFF_ALO = OFF_AHI + SZ_APK;
constexpr size_t OFF_BHI = OFF_ALO + SZ_APK;
constexpr size_t SZ_BPK  = (size_t)K2 * NFFT * 2;              // 2.16 MB
constexpr size_t OFF_BLO = OFF_BHI + SZ_BPK;
constexpr size_t OFF_FR  = OFF_BLO + SZ_BPK;                   // frames 67 MB

__device__ __forceinline__ void split_bf16(float x, __bf16& hi, __bf16& lo) {
    __bf16 h = (__bf16)x;
    hi = h;
    lo = (__bf16)(x - (float)h);
}

#if USE_TDM
__device__ __forceinline__ unsigned lds_addr_of(const void* p) {
    return (unsigned)(unsigned long long)
        (__attribute__((address_space(3))) const void*)p;
}

// 1-D TDM copy: n8 elements of 8 bytes, global -> LDS (D# per ISA 8.3/8.4)
__device__ __forceinline__ void tdm_load_1d(unsigned lds_byte, const void* gptr,
                                            unsigned n8) {
    unsigned long long ga = (unsigned long long)gptr;
    v4u g0;
    g0[0] = 1u;                                   // count=1 (valid descriptor)
    g0[1] = lds_byte;                             // lds_addr
    g0[2] = (unsigned)ga;                         // global_addr[31:0]
    g0[3] = (unsigned)(ga >> 32) | 0x80000000u;   // global_addr[56:32] | type=2
    v8i g1;
    g1[0] = (int)(3u << 16);                      // data_size=8B, mask=0
    g1[1] = (int)((n8 & 0xFFFFu) << 16);          // tensor_dim0[15:0]
    g1[2] = (int)(n8 >> 16);                      // tensor_dim0[31:16]
    g1[3] = (int)((n8 & 0xFFFFu) << 16);          // tile_dim0 = n8
    g1[4] = 0;                                    // tile_dim1/2 unused
    g1[5] = (int)n8;                              // tensor_dim0_stride
    g1[6] = 0;
    g1[7] = 0;
    v4i gz4 = {0, 0, 0, 0};
    v8i gz8 = {0, 0, 0, 0, 0, 0, 0, 0};
    __builtin_amdgcn_tensor_load_to_lds(g0, g1, gz4, gz4, gz8, 0);
}
#endif

// ------------------------- pack x into A fragments -------------------------
// A 16x32 bf16 layout: lane<16 -> row=lane, K = {kb..kb+7, kb+16..kb+23}, kb=0
//                      lane>=16 -> row=lane-16, kb=8
__global__ void pack_x_kernel(const float* __restrict__ x,
                              __bf16* __restrict__ xhi, __bf16* __restrict__ xlo) {
    int t    = blockIdx.x * 8 + (threadIdx.x >> 5);   // tile id: mtile*KT1 + kt
    int lane = threadIdx.x & 31;
    int mtile = t / KT1;
    int kt    = t % KT1;
    int row   = lane & 15;
    int kb    = (lane < 16) ? 0 : 8;
    const float* src = x + (size_t)(mtile * 16 + row) * DFEAT + kt * 32;
    v16bf hi, lo;
#pragma unroll
    for (int e = 0; e < 16; ++e) {
        int k = (e < 8) ? (kb + e) : (kb + 8 + e);   // kb+16+(e-8)
        __bf16 hv, lv;
        split_bf16(src[k], hv, lv);
        hi[e] = hv;
        lo[e] = lv;
    }
    size_t o = ((size_t)t * 32 + lane) * 16;
    *(v16bf*)(xhi + o) = hi;
    *(v16bf*)(xlo + o) = lo;
}

// ------------------------- pack W into B fragments -------------------------
// B 32x16 bf16 layout: lane<16 -> col=lane, K={0..7,16..23}; lane>=16 -> col=lane-16, K={8..15,24..31}
__global__ void pack_w_kernel(const float* __restrict__ W,
                              __bf16* __restrict__ whi, __bf16* __restrict__ wlo) {
    int t    = blockIdx.x * 8 + (threadIdx.x >> 5);   // tile id: kt*NT1 + nt
    int lane = threadIdx.x & 31;
    int kt  = t / NT1;
    int nt  = t % NT1;
    int col = nt * 16 + (lane & 15);
    int kb  = (lane < 16) ? 0 : 8;
    v16bf hi, lo;
#pragma unroll
    for (int e = 0; e < 16; ++e) {
        int krow = kt * 32 + ((e < 8) ? (kb + e) : (kb + 8 + e));
        float v = (col < OCOLS) ? W[(size_t)krow * OCOLS + col] : 0.0f;
        __bf16 hv, lv;
        split_bf16(v, hv, lv);
        hi[e] = hv;
        lo[e] = lv;
    }
    size_t o = ((size_t)t * 32 + lane) * 16;
    *(v16bf*)(whi + o) = hi;
    *(v16bf*)(wlo + o) = lo;
}

// --------------- windowed irfft basis, packed as B fragments ---------------
// Rows 0..512   : Cw[k][n] =  s_k/N * cos(2*pi*k*n/N) * win[n]
// Rows 513..1025: Dw[k][n] = -s_k/N * sin(2*pi*k*n/N) * win[n]
// Rows 1026..1055: zero padding.  s_k = 1 for k in {0,512}, else 2.
__global__ void pack_basis_kernel(__bf16* __restrict__ bhi, __bf16* __restrict__ blo) {
    int t    = blockIdx.x * 8 + (threadIdx.x >> 5);   // tile id: kt2*NT2 + nt
    int lane = threadIdx.x & 31;
    int kt2 = t / NT2;
    int nt  = t % NT2;
    int n   = nt * 16 + (lane & 15);
    int kb  = (lane < 16) ? 0 : 8;
    float win = 0.5f * (1.0f - __cosf((float)n * TWO_PI_OVER_N));
    v16bf hi, lo;
#pragma unroll
    for (int e = 0; e < 16; ++e) {
        int krow = kt2 * 32 + ((e < 8) ? (kb + e) : (kb + 8 + e));
        float v = 0.0f;
        if (krow < OCOLS) {
            int k = (krow < NBINS) ? krow : (krow - NBINS);
            float s = (k == 0 || k == NFFT / 2) ? 1.0f : 2.0f;
            int ph = (k * n) & (NFFT - 1);            // (k*n) mod 1024
            float sn, cs;
            __sincosf((float)ph * TWO_PI_OVER_N, &sn, &cs);
            float base = (krow < NBINS) ? cs : -sn;
            v = s * (1.0f / (float)NFFT) * base * win;
        }
        __bf16 hv, lv;
        split_bf16(v, hv, lv);
        hi[e] = hv;
        lo[e] = lv;
    }
    size_t o = ((size_t)t * 32 + lane) * 16;
    *(v16bf*)(bhi + o) = hi;
    *(v16bf*)(blo + o) = lo;
}

// ------------------- bf16x3 WMMA GEMM (16x64 per wave) ---------------------
// Per block: 8 waves share one 64-wide N stripe; the 4 B fragment pairs for
// the current k-tile (8 KB) are staged into LDS once per block via the Tensor
// Data Mover, then all waves read them with ds_load_b128.
__global__ void wmma_gemm_bf16x3(const __bf16* __restrict__ Ahi, const __bf16* __restrict__ Alo,
                                 const __bf16* __restrict__ Bhi, const __bf16* __restrict__ Blo,
                                 float* __restrict__ Out, const float* __restrict__ bias,
                                 int KT, int NT, int outStride, int validN) {
    __shared__ __align__(16) unsigned char smem[8192];   // [0,4K)=Bhi, [4K,8K)=Blo

    int lane  = threadIdx.x & 31;
    int wv    = threadIdx.x >> 5;
    int mtile = blockIdx.x * 8 + wv;
    int nt0   = blockIdx.y * 4;

    v8f acc[4] = {};
    const size_t abase = (size_t)mtile * KT;

    for (int kt = 0; kt < KT; ++kt) {
        // ---- stage 4 B fragment pairs for this k-tile into LDS ----
        size_t boff = ((size_t)kt * NT + nt0) * 512;     // __bf16 elements
#if USE_TDM
        if (wv == 0) {
            tdm_load_1d(lds_addr_of(smem),        Bhi + boff, 512);
            tdm_load_1d(lds_addr_of(smem) + 4096, Blo + boff, 512);
            __builtin_amdgcn_s_wait_tensorcnt(0);
        }
#else
        {   // cooperative fallback: 256 threads x 32B
            const uint4* gh = (const uint4*)(Bhi + boff);
            const uint4* gl = (const uint4*)(Blo + boff);
            uint4* sh = (uint4*)smem;
            sh[threadIdx.x]       = gh[threadIdx.x];
            sh[256 + threadIdx.x] = gl[threadIdx.x];
        }
#endif
        __syncthreads();

        size_t ao = ((abase + kt) * 32 + lane) * 16;
        const v16bf ahi = *(const v16bf*)(Ahi + ao);
        const v16bf alo = *(const v16bf*)(Alo + ao);
        if (kt + 1 < KT) {  // gfx1250 global_prefetch_b8 on next A fragment
            __builtin_prefetch(Ahi + ((abase + kt + 1) * 32 + lane) * 16, 0, 3);
            __builtin_prefetch(Alo + ((abase + kt + 1) * 32 + lane) * 16, 0, 3);
        }
#pragma unroll
        for (int c = 0; c < 4; ++c) {
            int nt = nt0 + c;
            if (nt >= NT) break;
            const v16bf bhi = *(const v16bf*)(smem + (c * 1024 + lane * 32));
            const v16bf blo = *(const v16bf*)(smem + (4096 + c * 1024 + lane * 32));
            // bf16x3: C += Ahi*Bhi + Ahi*Blo + Alo*Bhi  (~fp32 accuracy)
            acc[c] = __builtin_amdgcn_wmma_f32_16x16x32_bf16(
                false, ahi, false, bhi, (short)0, acc[c], false, false);
            acc[c] = __builtin_amdgcn_wmma_f32_16x16x32_bf16(
                false, ahi, false, blo, (short)0, acc[c], false, false);
            acc[c] = __builtin_amdgcn_wmma_f32_16x16x32_bf16(
                false, alo, false, bhi, (short)0, acc[c], false, false);
        }
        __syncthreads();   // protect smem before next k-tile's DMA
    }

    // C/D layout: VGPR r -> row = r + (lane<16 ? 0 : 8), col = lane%16
    int row0 = (lane >> 4) << 3;
    int col  = lane & 15;
#pragma unroll
    for (int c = 0; c < 4; ++c) {
        int nt = nt0 + c;
        if (nt >= NT) break;
        int gcol = nt * 16 + col;
        if (gcol >= validN) continue;
        float bv = bias ? bias[gcol] : 0.0f;
#pragma unroll
        for (int r = 0; r < 8; ++r) {
            int grow = mtile * 16 + row0 + r;
            Out[(size_t)grow * outStride + gcol] = acc[c][r] + bv;
        }
    }
}

// -------- activation: mag/phase -> [mag*cos(p) | mag*sin(p)], A-pack -------
__global__ void act_pack_kernel(const float* __restrict__ h,
                                __bf16* __restrict__ ahi, __bf16* __restrict__ alo) {
    int t    = blockIdx.x * 8 + (threadIdx.x >> 5);   // tile id: mtile*KT2 + kt2
    int lane = threadIdx.x & 31;
    int mtile = t / KT2;
    int kt2   = t % KT2;
    int row   = lane & 15;
    int kb    = (lane < 16) ? 0 : 8;
    const float* hrow = h + (size_t)(mtile * 16 + row) * NP1;
    v16bf hi, lo;
#pragma unroll
    for (int e = 0; e < 16; ++e) {
        int k2 = kt2 * 32 + ((e < 8) ? (kb + e) : (kb + 8 + e));
        float v = 0.0f;
        if (k2 < OCOLS) {
            int k    = (k2 < NBINS) ? k2 : (k2 - NBINS);
            float mg = fminf(__expf(hrow[k]), 100.0f);
            float p  = hrow[NBINS + k];
            float sn, cs;
            __sincosf(p, &sn, &cs);
            v = mg * ((k2 < NBINS) ? cs : sn);
        }
        __bf16 hv, lv;
        split_bf16(v, hv, lv);
        hi[e] = hv;
        lo[e] = lv;
    }
    size_t o = ((size_t)t * 32 + lane) * 16;
    *(v16bf*)(ahi + o) = hi;
    *(v16bf*)(alo + o) = lo;
}

// ---------------- gather overlap-add + analytic envelope -------------------
__global__ void overlap_add_kernel(const float* __restrict__ frames,
                                   float* __restrict__ out) {
    int idx = blockIdx.x * blockDim.x + threadIdx.x;   // 0 .. 8*524288-1
    int b  = idx >> 19;
    int m  = idx & (OUTLEN - 1);
    int gm = m + PADC;
    int tmax = gm >> 8;
    if (tmax > LFRAMES - 1) tmax = LFRAMES - 1;
    int tmin = (gm - (NFFT - 1) + (HOP - 1)) >> 8;     // ceil((gm-1023)/256)
    if (tmin < 0) tmin = 0;
    float acc = 0.0f, env = 0.0f;
    for (int t = tmin; t <= tmax; ++t) {
        int off = gm - (t << 8);
        float w = 0.5f * (1.0f - __cosf((float)off * TWO_PI_OVER_N));
        acc += frames[(((size_t)b * LFRAMES + t) << 10) + off];
        env += w * w;
    }
    out[idx] = acc / env;
}

// ---------------------------------------------------------------------------
extern "C" void kernel_launch(void* const* d_in, const int* in_sizes, int n_in,
                              void* d_out, int out_size, void* d_ws, size_t ws_size,
                              hipStream_t stream) {
    (void)in_sizes; (void)n_in; (void)out_size; (void)ws_size;
    const float* x    = (const float*)d_in[0];
    const float* W    = (const float*)d_in[1];
    const float* bias = (const float*)d_in[2];
    float*       out  = (float*)d_out;
    char*        ws   = (char*)d_ws;

    __bf16* xhi = (__bf16*)(ws + OFF_XHI);
    __bf16* xlo = (__bf16*)(ws + OFF_XLO);
    __bf16* whi = (__bf16*)(ws + OFF_WHI);
    __bf16* wlo = (__bf16*)(ws + OFF_WLO);
    float*  h   = (float*)(ws + OFF_H);
    __bf16* ahi = (__bf16*)(ws + OFF_AHI);
    __bf16* alo = (__bf16*)(ws + OFF_ALO);
    __bf16* bhi = (__bf16*)(ws + OFF_BHI);
    __bf16* blo = (__bf16*)(ws + OFF_BLO);
    float*  fr  = (float*)(ws + OFF_FR);

    // 1) pack operands (basis generated analytically, window/scaling folded in)
    pack_basis_kernel<<<(KT2 * NT2) / 8, 256, 0, stream>>>(bhi, blo);
    pack_w_kernel<<<(KT1 * NT1) / 8, 256, 0, stream>>>(W, whi, wlo);
    pack_x_kernel<<<(MT * KT1) / 8, 256, 0, stream>>>(x, xhi, xlo);

    // 2) GEMM1: h = x @ W + b   (M=16384, K=512, N=1026 padded to 1040)
    wmma_gemm_bf16x3<<<dim3(MT / 8, (NT1 + 3) / 4), 256, 0, stream>>>(
        xhi, xlo, whi, wlo, h, bias, KT1, NT1, NP1, OCOLS);

    // 3) activation + repack as A fragments (K padded 1026 -> 1056)
    act_pack_kernel<<<(MT * KT2) / 8, 256, 0, stream>>>(h, ahi, alo);

    // 4) GEMM2: frames = a2 @ basis  (M=16384, K=1056, N=1024)
    wmma_gemm_bf16x3<<<dim3(MT / 8, NT2 / 4), 256, 0, stream>>>(
        ahi, alo, bhi, blo, fr, nullptr, KT2, NT2, NFFT, NFFT);

    // 5) overlap-add gather + envelope division
    overlap_add_kernel<<<(BATCH * OUTLEN) / 256, 256, 0, stream>>>(fr, out);
}
